// SlotAttentionAdjusted_54571854463153
// MI455X (gfx1250) — compile-verified, matrix operand order checked by reading, not run on hardware
//
#include <hip/hip_runtime.h>
#include <hip/hip_bf16.h>
#include <stdint.h>

// ---------------------------------------------------------------------------
// Problem constants (match reference)
// ---------------------------------------------------------------------------
#define B_   32
#define R_   4096
#define D_   512
#define S_   8
#define SP_  16            // slots padded to 16 for WMMA M dimension
#define H_   512
#define SCALE_ 0.04419417382415922f   // 512^-0.5
#define LN_EPS_ 1e-5f
#define EPS_    1e-8f

typedef __attribute__((ext_vector_type(16))) __bf16 v16bf;
typedef __attribute__((ext_vector_type(8)))  float  v8f;

// ---------------------------------------------------------------------------
// WMMA helpers
// ---------------------------------------------------------------------------
static __device__ __forceinline__ v8f wmma_bf16(v16bf a, v16bf b, v8f c) {
  return __builtin_amdgcn_wmma_f32_16x16x32_bf16(false, a, false, b, (short)0, c,
                                                 false, false);
}

// Load a 16-element (per lane) bf16 fragment following the CDNA5 16-bit matrix
// striping: lanes 0-15 hold K=0..7 and 16..23, lanes 16-31 hold K=8..15 and
// 24..31 (kh = lane>>4 selects the half).  p must point at (row, k0) with the
// K axis contiguous; 16B aligned (k0 multiple of 32, row stride multiple of 32).
static __device__ __forceinline__ v16bf load_frag_kcontig(const __bf16* p, int kh) {
  const uint4* q = (const uint4*)p;       // q[0]:K0-7 q[1]:K8-15 q[2]:K16-23 q[3]:K24-31
  union { uint4 u[2]; v16bf v; } t;
  t.u[0] = q[kh];
  t.u[1] = q[2 + kh];
  return t.v;
}

// ---------------------------------------------------------------------------
// Block reduction (256 threads, wave32)
// ---------------------------------------------------------------------------
static __device__ __forceinline__ float block_sum256(float v) {
  __shared__ float red[8];
#pragma unroll
  for (int o = 16; o; o >>= 1) v += __shfl_xor(v, o, 32);
  const int lane = threadIdx.x & 31, w = threadIdx.x >> 5;
  __syncthreads();
  if (lane == 0) red[w] = v;
  __syncthreads();
  float t = red[threadIdx.x & 7];
#pragma unroll
  for (int o = 4; o; o >>= 1) t += __shfl_xor(t, o, 32);
  return t;   // every thread holds the full sum
}

// ---------------------------------------------------------------------------
// Small utility kernels
// ---------------------------------------------------------------------------
__global__ void k_f32_to_bf16(const float* __restrict__ src, __bf16* __restrict__ dst,
                              int n) {
  int i = blockIdx.x * 256 + threadIdx.x;
  if (i < n) dst[i] = (__bf16)src[i];
}

__global__ void k_zero256(float* __restrict__ p) { p[threadIdx.x] = 0.f; }

__global__ void k_init_slots(const float* __restrict__ slots_init,
                             float* __restrict__ slots) {
  size_t idx = (size_t)blockIdx.x * 256 + threadIdx.x;   // B*S*D
  slots[idx] = slots_init[idx % (size_t)(S_ * D_)];
}

// ---------------------------------------------------------------------------
// LayerNorm over 512-wide rows -> bf16 (one row per 256-thread block)
// ---------------------------------------------------------------------------
__global__ void k_ln_rows_bf16(const float* __restrict__ x, const float* __restrict__ w,
                               const float* __restrict__ b, __bf16* __restrict__ y) {
  const size_t row = blockIdx.x;
  const float* xr = x + row * D_;
  const float v0 = xr[threadIdx.x], v1 = xr[threadIdx.x + 256];
  const float mu = block_sum256(v0 + v1) * (1.f / D_);
  const float d0 = v0 - mu, d1 = v1 - mu;
  const float var = block_sum256(d0 * d0 + d1 * d1) * (1.f / D_);
  const float rs = rsqrtf(var + LN_EPS_);
  __bf16* yr = y + row * D_;
  yr[threadIdx.x]       = (__bf16)(d0 * rs * w[threadIdx.x] + b[threadIdx.x]);
  yr[threadIdx.x + 256] = (__bf16)(d1 * rs * w[threadIdx.x + 256] + b[threadIdx.x + 256]);
}

// Slot layernorm with padding: rows = B*16; pad rows (i>=8) written as zero.
__global__ void k_ln_slots_pad(const float* __restrict__ slots,
                               const float* __restrict__ w, const float* __restrict__ b,
                               __bf16* __restrict__ y) {
  const int row = blockIdx.x;              // b*16 + i
  const int bi = row >> 4, i = row & 15;
  __bf16* yr = y + (size_t)row * D_;
  if (i >= S_) {
    yr[threadIdx.x] = (__bf16)0.f;
    yr[threadIdx.x + 256] = (__bf16)0.f;
    return;
  }
  const float* xr = slots + ((size_t)bi * S_ + i) * D_;
  const float v0 = xr[threadIdx.x], v1 = xr[threadIdx.x + 256];
  const float mu = block_sum256(v0 + v1) * (1.f / D_);
  const float d0 = v0 - mu, d1 = v1 - mu;
  const float var = block_sum256(d0 * d0 + d1 * d1) * (1.f / D_);
  const float rs = rsqrtf(var + LN_EPS_);
  yr[threadIdx.x]       = (__bf16)(d0 * rs * w[threadIdx.x] + b[threadIdx.x]);
  yr[threadIdx.x + 256] = (__bf16)(d1 * rs * w[threadIdx.x + 256] + b[threadIdx.x + 256]);
}

// ---------------------------------------------------------------------------
// GEMM:  out[r, c] = sum_d A[r, d] * W[c, d] + bias[c]     (K = 512, NT form)
// grid.x = rows/16, grid.y = cols/128; 8 waves/block, one 16x16 tile per wave.
// Used for k = LN(x) @ Wk^T and q = LN(slots) @ Wq^T.
// ---------------------------------------------------------------------------
__global__ void k_gemm_nt(const __bf16* __restrict__ A, const __bf16* __restrict__ W,
                          const float* __restrict__ bias, __bf16* __restrict__ out) {
  const int lane = threadIdx.x & 31, wave = threadIdx.x >> 5;
  const int row0 = blockIdx.x * 16;
  const int col0 = blockIdx.y * 128 + wave * 16;
  const int m = lane & 15, kh = lane >> 4;
  const __bf16* ap = A + (size_t)(row0 + m) * D_;
  const __bf16* bp = W + (size_t)(col0 + m) * D_;
  v8f acc = {};
#pragma unroll
  for (int k0 = 0; k0 < D_; k0 += 32) {
    v16bf a = load_frag_kcontig(ap + k0, kh);
    v16bf b = load_frag_kcontig(bp + k0, kh);
    acc = wmma_bf16(a, b, acc);
  }
  // C/D layout: VGPR i -> M = i + 8*(lane>=16); N = lane&15
  const float bi = bias[col0 + m];
  __bf16* op = out + (size_t)(row0 + kh * 8) * D_ + col0 + m;
#pragma unroll
  for (int i = 0; i < 8; ++i) op[(size_t)i * D_] = (__bf16)(acc[i] + bi);
}

// ---------------------------------------------------------------------------
// Transposed v projection: vT[b, d, j] = sum_kd Wv[d, kd] * x[b*R + j, kd] + bv[d]
// A = Wv rows (M = d), B = x rows (N = j) -> same NT fragment pattern, but the
// output is written feature-major so the updates GEMM can read it K-contiguous.
// grid.x = D/16, grid.y = BR/128.
// ---------------------------------------------------------------------------
__global__ void k_gemm_vT(const __bf16* __restrict__ Wv, const __bf16* __restrict__ x,
                          const float* __restrict__ bias, __bf16* __restrict__ vT) {
  const int lane = threadIdx.x & 31, wave = threadIdx.x >> 5;
  const int d0 = blockIdx.x * 16;
  const int n0 = blockIdx.y * 128 + wave * 16;   // global row index b*R + j
  const int m = lane & 15, kh = lane >> 4;
  const __bf16* ap = Wv + (size_t)(d0 + m) * D_;
  const __bf16* bp = x + (size_t)(n0 + m) * D_;
  v8f acc = {};
#pragma unroll
  for (int k0 = 0; k0 < D_; k0 += 32) {
    v16bf a = load_frag_kcontig(ap + k0, kh);
    v16bf b = load_frag_kcontig(bp + k0, kh);
    acc = wmma_bf16(a, b, acc);
  }
  const int batch = n0 >> 12;        // n0 / R_
  const int jj = n0 & (R_ - 1);      // n0 % R_
  __bf16* op = vT + (size_t)batch * D_ * R_ + (size_t)(d0 + kh * 8) * R_ + jj + m;
#pragma unroll
  for (int i = 0; i < 8; ++i)
    op[(size_t)i * R_] = (__bf16)(acc[i] + bias[d0 + kh * 8 + i]);
}

// ---------------------------------------------------------------------------
// dots[b, i, j] = SCALE * sum_d q[b, i, d] * k[b, j, d]   (i < 8 stored)
// grid = (R/128, B)
// ---------------------------------------------------------------------------
__global__ void k_dots(const __bf16* __restrict__ q, const __bf16* __restrict__ kk,
                       float* __restrict__ dots) {
  const int lane = threadIdx.x & 31, wave = threadIdx.x >> 5;
  const int b = blockIdx.y;
  const int j0 = blockIdx.x * 128 + wave * 16;
  const int m = lane & 15, kh = lane >> 4;
  const __bf16* ap = q + ((size_t)b * SP_ + m) * D_;
  const __bf16* bp = kk + ((size_t)b * R_ + j0 + m) * D_;
  v8f acc = {};
#pragma unroll
  for (int k0 = 0; k0 < D_; k0 += 32) {
    v16bf a = load_frag_kcontig(ap + k0, kh);
    v16bf bt = load_frag_kcontig(bp + k0, kh);
    acc = wmma_bf16(a, bt, acc);
  }
  if (kh == 0) {   // rows M=0..7 are the real slots; M=8..15 are padding
#pragma unroll
    for (int i = 0; i < 8; ++i)
      dots[((size_t)b * S_ + i) * R_ + j0 + m] = acc[i] * SCALE_;
  }
}

// ---------------------------------------------------------------------------
// Softmax over slots (axis=1) per (b, j) column.  Writes UNNORMALIZED
// (softmax + EPS) attention directly as padded bf16 [B, 16, R] and accumulates
// the per-(b,i) R-sums; the division by rowsum is folded into k_updates.
// grid = (R/256, B)
// ---------------------------------------------------------------------------
__global__ void k_softmax_slots(const float* __restrict__ dots,
                                __bf16* __restrict__ attn_bf,
                                float* __restrict__ rowsum) {
  const int b = blockIdx.y;
  const int j = blockIdx.x * 256 + threadIdx.x;
  const float* dp = dots + (size_t)b * S_ * R_ + j;
  float x[S_], mx = -3.4e38f;
#pragma unroll
  for (int i = 0; i < S_; ++i) { x[i] = dp[(size_t)i * R_]; mx = fmaxf(mx, x[i]); }
  float s = 0.f;
#pragma unroll
  for (int i = 0; i < S_; ++i) { x[i] = __expf(x[i] - mx); s += x[i]; }
  const float inv = 1.f / s;
  __bf16* ap = attn_bf + (size_t)b * SP_ * R_ + j;
  float part[S_];
#pragma unroll
  for (int i = 0; i < S_; ++i) {
    const float a = x[i] * inv + EPS_;
    ap[(size_t)i * R_] = (__bf16)a;
    part[i] = a;
  }
#pragma unroll
  for (int i = S_; i < SP_; ++i) ap[(size_t)i * R_] = (__bf16)0.f;  // pad rows
#pragma unroll
  for (int i = 0; i < S_; ++i) {
    const float t = block_sum256(part[i]);
    if (threadIdx.x == 0) atomicAdd(&rowsum[b * S_ + i], t);
  }
}

// ---------------------------------------------------------------------------
// updates[b, i, d] = (sum_j attnU[b, i, j] * vT[b, d, j]) / rowsum[b, i]
// A = attnU rows (K = j contiguous), B = vT rows (lane n = d, K = j contiguous)
// -> pure b128-load fragments.  grid = (D/128, B), K = R = 4096.
// ---------------------------------------------------------------------------
__global__ void k_updates(const __bf16* __restrict__ attn_bf,
                          const __bf16* __restrict__ vT,
                          const float* __restrict__ rowsum,
                          float* __restrict__ upd) {
  const int lane = threadIdx.x & 31, wave = threadIdx.x >> 5;
  const int b = blockIdx.y;
  const int d0 = blockIdx.x * 128 + wave * 16;
  const int m = lane & 15, kh = lane >> 4;
  const __bf16* ap = attn_bf + ((size_t)b * SP_ + m) * R_;
  const __bf16* bp = vT + (size_t)b * D_ * R_ + (size_t)(d0 + m) * R_;
  v8f acc = {};
  for (int k0 = 0; k0 < R_; k0 += 32) {
    __builtin_prefetch(ap + k0 + 128, 0, 1);                 // global_prefetch_b8
    __builtin_prefetch(bp + k0 + 128, 0, 1);
    v16bf a = load_frag_kcontig(ap + k0, kh);
    v16bf bt = load_frag_kcontig(bp + k0, kh);
    acc = wmma_bf16(a, bt, acc);
  }
  if (kh == 0) {
#pragma unroll
    for (int i = 0; i < 8; ++i)
      upd[((size_t)b * S_ + i) * D_ + d0 + m] = acc[i] / rowsum[b * S_ + i];
  }
}

// ---------------------------------------------------------------------------
// GRU cell (PyTorch semantics), one row per block, f32 VALU (tiny workload).
// ---------------------------------------------------------------------------
__global__ void k_gru(const float* __restrict__ upd, const float* __restrict__ W_ih,
                      const float* __restrict__ b_ih, const float* __restrict__ W_hh,
                      const float* __restrict__ b_hh, float* __restrict__ slots) {
  __shared__ float xs[D_], hs[D_], gi[3 * D_], gh[3 * D_];
  const size_t row = blockIdx.x;
  const float* xr = upd + row * D_;
  float* hr = slots + row * D_;
  xs[threadIdx.x] = xr[threadIdx.x]; xs[threadIdx.x + 256] = xr[threadIdx.x + 256];
  hs[threadIdx.x] = hr[threadIdx.x]; hs[threadIdx.x + 256] = hr[threadIdx.x + 256];
  __syncthreads();
  for (int o = threadIdx.x; o < 3 * D_; o += 256) {
    const float* wi = W_ih + (size_t)o * D_;
    const float* wh = W_hh + (size_t)o * D_;
    float si = b_ih[o], sh = b_hh[o];
    for (int d = 0; d < D_; ++d) { si += wi[d] * xs[d]; sh += wh[d] * hs[d]; }
    gi[o] = si; gh[o] = sh;
  }
  __syncthreads();
  for (int d = threadIdx.x; d < D_; d += 256) {
    const float r = 1.f / (1.f + __expf(-(gi[d] + gh[d])));
    const float z = 1.f / (1.f + __expf(-(gi[D_ + d] + gh[D_ + d])));
    const float n = tanhf(gi[2 * D_ + d] + r * gh[2 * D_ + d]);
    hr[d] = (1.f - z) * n + z * hs[d];
  }
}

// ---------------------------------------------------------------------------
// Residual MLP with pre-LN, one row per block.  outbuf may alias slots or be
// the final d_out on the last iteration.
// ---------------------------------------------------------------------------
__global__ void k_mlp(const float* __restrict__ slots, const float* __restrict__ lnw,
                      const float* __restrict__ lnb, const float* __restrict__ W1,
                      const float* __restrict__ b1, const float* __restrict__ W2,
                      const float* __restrict__ b2, float* __restrict__ outbuf) {
  __shared__ float pre[D_], hid[H_];
  const size_t row = blockIdx.x;
  const float* sr = slots + row * D_;
  const float v0 = sr[threadIdx.x], v1 = sr[threadIdx.x + 256];
  const float mu = block_sum256(v0 + v1) * (1.f / D_);
  const float d0 = v0 - mu, d1 = v1 - mu;
  const float var = block_sum256(d0 * d0 + d1 * d1) * (1.f / D_);
  const float rs = rsqrtf(var + LN_EPS_);
  pre[threadIdx.x]       = d0 * rs * lnw[threadIdx.x] + lnb[threadIdx.x];
  pre[threadIdx.x + 256] = d1 * rs * lnw[threadIdx.x + 256] + lnb[threadIdx.x + 256];
  __syncthreads();
  for (int o = threadIdx.x; o < H_; o += 256) {
    const float* w = W1 + (size_t)o * D_;
    float s = b1[o];
    for (int d = 0; d < D_; ++d) s += w[d] * pre[d];
    hid[o] = fmaxf(s, 0.f);
  }
  __syncthreads();
  for (int o = threadIdx.x; o < D_; o += 256) {
    const float* w = W2 + (size_t)o * H_;
    float s = b2[o];
    for (int d = 0; d < H_; ++d) s += w[d] * hid[d];
    outbuf[row * D_ + o] = sr[o] + s;
  }
}

// ---------------------------------------------------------------------------
// Host-side orchestration
// ---------------------------------------------------------------------------
extern "C" void kernel_launch(void* const* d_in, const int* in_sizes, int n_in,
                              void* d_out, int out_size, void* d_ws, size_t ws_size,
                              hipStream_t stream) {
  const float* inputs     = (const float*)d_in[0];
  const float* slots_init = (const float*)d_in[1];
  const float* Wq = (const float*)d_in[2];  const float* bq = (const float*)d_in[3];
  const float* Wk = (const float*)d_in[4];  const float* bk = (const float*)d_in[5];
  const float* Wv = (const float*)d_in[6];  const float* bv = (const float*)d_in[7];
  const float* W_ih = (const float*)d_in[8];  const float* b_ih = (const float*)d_in[9];
  const float* W_hh = (const float*)d_in[10]; const float* b_hh = (const float*)d_in[11];
  const float* W1 = (const float*)d_in[12]; const float* b1 = (const float*)d_in[13];
  const float* W2 = (const float*)d_in[14]; const float* b2 = (const float*)d_in[15];
  const float* ln_in_w = (const float*)d_in[16]; const float* ln_in_b = (const float*)d_in[17];
  const float* ln_s_w  = (const float*)d_in[18]; const float* ln_s_b  = (const float*)d_in[19];
  const float* ln_ff_w = (const float*)d_in[20]; const float* ln_ff_b = (const float*)d_in[21];

  // Workspace carve-up
  char* wp = (char*)d_ws;
  auto alloc = [&](size_t bytes) {
    void* p = (void*)wp;
    wp += (bytes + 255) & ~(size_t)255;
    return p;
  };
  const size_t BR = (size_t)B_ * R_;
  __bf16* x_bf    = (__bf16*)alloc(BR * D_ * 2);
  __bf16* k_bf    = (__bf16*)alloc(BR * D_ * 2);
  __bf16* vT_bf   = (__bf16*)alloc(BR * D_ * 2);           // [B, D, R]
  __bf16* wq_bf   = (__bf16*)alloc((size_t)D_ * D_ * 2);
  __bf16* wk_bf   = (__bf16*)alloc((size_t)D_ * D_ * 2);
  __bf16* wv_bf   = (__bf16*)alloc((size_t)D_ * D_ * 2);
  __bf16* sln_bf  = (__bf16*)alloc((size_t)B_ * SP_ * D_ * 2);
  __bf16* q_bf    = (__bf16*)alloc((size_t)B_ * SP_ * D_ * 2);
  float*  dots    = (float*) alloc((size_t)B_ * S_ * R_ * 4);
  __bf16* attn_bf = (__bf16*)alloc((size_t)B_ * SP_ * R_ * 2);
  float*  rowsum  = (float*) alloc((size_t)B_ * S_ * 4);
  float*  upd     = (float*) alloc((size_t)B_ * S_ * D_ * 4);
  float*  slots   = (float*) alloc((size_t)B_ * S_ * D_ * 4);

  const dim3 blk(256);

  // Weights for the WMMA GEMMs -> bf16
  k_f32_to_bf16<<<(D_ * D_ + 255) / 256, blk, 0, stream>>>(Wq, wq_bf, D_ * D_);
  k_f32_to_bf16<<<(D_ * D_ + 255) / 256, blk, 0, stream>>>(Wk, wk_bf, D_ * D_);
  k_f32_to_bf16<<<(D_ * D_ + 255) / 256, blk, 0, stream>>>(Wv, wv_bf, D_ * D_);

  // x = LN(inputs) in bf16; then projections (the two big GEMMs)
  k_ln_rows_bf16<<<(unsigned)BR, blk, 0, stream>>>(inputs, ln_in_w, ln_in_b, x_bf);
  k_gemm_nt<<<dim3(BR / 16, D_ / 128), blk, 0, stream>>>(x_bf, wk_bf, bk, k_bf);
  k_gemm_vT<<<dim3(D_ / 16, BR / 128), blk, 0, stream>>>(wv_bf, x_bf, bv, vT_bf);

  // slots = broadcast(slots_init)
  k_init_slots<<<(B_ * S_ * D_) / 256, blk, 0, stream>>>(slots_init, slots);

  for (int it = 0; it < 3; ++it) {
    k_ln_slots_pad<<<B_ * SP_, blk, 0, stream>>>(slots, ln_s_w, ln_s_b, sln_bf);
    k_gemm_nt<<<dim3((B_ * SP_) / 16, D_ / 128), blk, 0, stream>>>(sln_bf, wq_bf, bq, q_bf);
    k_dots<<<dim3(R_ / 128, B_), blk, 0, stream>>>(q_bf, k_bf, dots);
    k_zero256<<<1, blk, 0, stream>>>(rowsum);
    k_softmax_slots<<<dim3(R_ / 256, B_), blk, 0, stream>>>(dots, attn_bf, rowsum);
    k_updates<<<dim3(D_ / 128, B_), blk, 0, stream>>>(attn_bf, vT_bf, rowsum, upd);
    k_gru<<<B_ * S_, blk, 0, stream>>>(upd, W_ih, b_ih, W_hh, b_hh, slots);
    float* mlp_out = (it == 2) ? (float*)d_out : slots;
    k_mlp<<<B_ * S_, blk, 0, stream>>>(slots, ln_ff_w, ln_ff_b, W1, b1, W2, b2, mlp_out);
  }
}
